// TransGAT_88141318849066
// MI455X (gfx1250) — matrix operand order repaired and checked by reading, not compiled, not run on hardware
//
#include <hip/hip_runtime.h>
#include <stdint.h>

// ---- problem constants (match reference) ----
#define NNODES 32768
#define NB     64
#define NS     512
#define NE     524288
#define NDIN   128
#define NDOUT  128
#define NEDIM  16
#define NH     8
#define NDH    16
#define ETOT   (NE + NNODES)   // edges + self loops

typedef __attribute__((ext_vector_type(16))) __bf16          v16bf;
typedef __attribute__((ext_vector_type(8)))  float           v8f;
typedef __attribute__((ext_vector_type(8)))  unsigned short  u16x8;

// ---------- helpers ----------
__device__ __forceinline__ unsigned short f2bf(float f) {
    unsigned u = __float_as_uint(f);
    unsigned r = 0x7FFFu + ((u >> 16) & 1u);
    return (unsigned short)((u + r) >> 16);
}
__device__ __forceinline__ unsigned enc_ord(float f) {
    unsigned u = __float_as_uint(f);
    return (u & 0x80000000u) ? ~u : (u | 0x80000000u);
}
__device__ __forceinline__ float dec_ord(unsigned e) {
    unsigned u = (e & 0x80000000u) ? (e & 0x7FFFFFFFu) : ~e;
    return __uint_as_float(u);
}

// ---------- small utility kernels ----------
__global__ void k_zero_f(float* p, int n) {
    int i = blockIdx.x * blockDim.x + threadIdx.x;
    if (i < n) p[i] = 0.0f;
}
__global__ void k_zero_u(unsigned* p, int n) {
    int i = blockIdx.x * blockDim.x + threadIdx.x;
    if (i < n) p[i] = 0u;
}
// straight f32 -> bf16
__global__ void k_cvt_bf(const float* __restrict__ in, unsigned short* __restrict__ out, int n) {
    int i = blockIdx.x * blockDim.x + threadIdx.x;
    if (i < n) out[i] = f2bf(in[i]);
}
// W [K, Ncol] f32 -> Bt [Ncol, Kpad] bf16 (transpose + zero pad)
__global__ void k_cvt_t(const float* __restrict__ w, unsigned short* __restrict__ bt,
                        int K, int Ncol, int Kpad) {
    int i = blockIdx.x * blockDim.x + threadIdx.x;
    int n = i / Kpad, k = i - n * Kpad;
    if (n >= Ncol) return;
    bt[i] = (k < K) ? f2bf(w[k * Ncol + n]) : (unsigned short)0;
}

// ---------- per-graph max pool ----------
__global__ void k_pool_max(const float* __restrict__ x, float* __restrict__ gc) {
    int g = blockIdx.x, c = threadIdx.x;
    const float* p = x + (size_t)g * NS * NDIN + c;
    float m = -3.4e38f;
    for (int s = 0; s < NS; ++s) m = fmaxf(m, p[(size_t)s * NDIN]);
    gc[g * NDIN + c] = m;
}

// build x_cat (bf16 [N,256]) and x (bf16 [N,128])
__global__ void k_build_xcat(const float* __restrict__ x, const float* __restrict__ gc,
                             const int* __restrict__ batch,
                             unsigned short* __restrict__ xc, unsigned short* __restrict__ xb) {
    int i = blockIdx.x * blockDim.x + threadIdx.x;   // over N*256
    int node = i >> 8, c = i & 255;
    float v;
    if (c < NDIN) { v = x[node * NDIN + c]; xb[node * NDIN + c] = f2bf(v); }
    else          { v = gc[batch[node] * NDIN + (c - NDIN)]; }
    xc[i] = f2bf(v);
}

// ---------- WMMA GEMM: C[M,128] = A_bf16[M,K] @ Bt_bf16[128,K] + bias ----------
// block = 256 threads = 8 waves; each wave owns one 16x16 output tile; grid.x = M/16
__global__ __launch_bounds__(256)
void k_gemm_bf16(const unsigned short* __restrict__ A, int lda,
                 const unsigned short* __restrict__ Bt, int ldb,
                 const float* __restrict__ bias,
                 float* __restrict__ C, int ldc, int K) {
    const int wave = threadIdx.x >> 5;
    const int lane = threadIdx.x & 31;
    const int half = lane >> 4;
    const int l16  = lane & 15;
    const int row0 = blockIdx.x * 16;
    const int col0 = wave * 16;
    const int kb0  = half * 8;

    union { v16bf v; u16x8 h[2]; } af, bfr;
    v8f acc = {};
    const unsigned short* ap = A  + (size_t)(row0 + l16) * lda;
    const unsigned short* bp = Bt + (size_t)(col0 + l16) * ldb + half * 16;
    for (int k0 = 0; k0 < K; k0 += 32) {
        af.h[0]  = *(const u16x8*)(ap + k0 + half * 16);
        af.h[1]  = *(const u16x8*)(ap + k0 + half * 16 + 8);
        bfr.h[0] = *(const u16x8*)(bp + k0);
        bfr.h[1] = *(const u16x8*)(bp + k0 + 8);
        acc = __builtin_amdgcn_wmma_f32_16x16x32_bf16(false, af.v, false, bfr.v,
                                                      (short)0, acc, false, false);
    }
    const int rbase = half * 8;
    float bsum = bias ? bias[col0 + l16] : 0.0f;
    float* cp = C + (size_t)(row0 + rbase) * ldc + col0 + l16;
#pragma unroll
    for (int r = 0; r < 8; ++r) cp[(size_t)r * ldc] = acc[r] + bsum;
}

// ---------- edge-attr mean ----------
__global__ void k_ea_sum(const float* __restrict__ ea, float* __restrict__ acc) {
    __shared__ float s[NEDIM];
    if (threadIdx.x < NEDIM) s[threadIdx.x] = 0.0f;
    __syncthreads();
    size_t stride = (size_t)gridDim.x * blockDim.x;
    for (size_t i = (size_t)blockIdx.x * blockDim.x + threadIdx.x;
         i < (size_t)NE * NEDIM; i += stride)
        atomicAdd(&s[i & (NEDIM - 1)], ea[i]);
    __syncthreads();
    if (threadIdx.x < NEDIM) atomicAdd(&acc[threadIdx.x], s[threadIdx.x]);
}
__global__ void k_eproj_mean(const float* __restrict__ acc, const float* __restrict__ We,
                             float* __restrict__ epm) {
    int c = threadIdx.x;
    float s = 0.0f;
    for (int t = 0; t < NEDIM; ++t) s += (acc[t] * (1.0f / (float)NE)) * We[t * NDOUT + c];
    epm[c] = s;
}

// ---------- edge passes (one wave32 per edge, 4 features per lane) ----------
__global__ void k_edge_logits(const int* __restrict__ src, const int* __restrict__ dst,
                              const float* __restrict__ ea, const float* __restrict__ We,
                              const float* __restrict__ epm,
                              const float* __restrict__ xl, const float* __restrict__ xr,
                              const float* __restrict__ att,
                              float* __restrict__ logit, unsigned* __restrict__ mx) {
    int e = (blockIdx.x * blockDim.x + threadIdx.x) >> 5;
    int lane = threadIdx.x & 31;
    if (e >= ETOT) return;
    int s, d, c = lane * 4;
    float ep[4];
    if (e < NE) {
        s = src[e]; d = dst[e];
        const float* eap = ea + (size_t)e * NEDIM;
        float ev[NEDIM];
#pragma unroll
        for (int t = 0; t < NEDIM; ++t) ev[t] = eap[t];
#pragma unroll
        for (int j = 0; j < 4; ++j) {
            float a = 0.0f;
#pragma unroll
            for (int t = 0; t < NEDIM; ++t) a += ev[t] * We[t * NDOUT + c + j];
            ep[j] = a;
        }
    } else {
        s = d = e - NE;
#pragma unroll
        for (int j = 0; j < 4; ++j) ep[j] = epm[c + j];
    }
    float4 lv = *(const float4*)(xl + (size_t)s * NDOUT + c);
    float4 rv = *(const float4*)(xr + (size_t)d * NDOUT + c);
    float m0 = lv.x + rv.x + ep[0], m1 = lv.y + rv.y + ep[1];
    float m2 = lv.z + rv.z + ep[2], m3 = lv.w + rv.w + ep[3];
    m0 = m0 > 0.f ? m0 : 0.2f * m0;  m1 = m1 > 0.f ? m1 : 0.2f * m1;
    m2 = m2 > 0.f ? m2 : 0.2f * m2;  m3 = m3 > 0.f ? m3 : 0.2f * m3;
    float a = m0 * att[c] + m1 * att[c + 1] + m2 * att[c + 2] + m3 * att[c + 3];
    for (int off = 16; off; off >>= 1) a += __shfl_xor(a, off, 32);
    if (lane == 0) {
        logit[e] = a;
        atomicMax(mx + d, enc_ord(a));
    }
}
__global__ void k_edge_w(const int* __restrict__ dst, const float* __restrict__ logit,
                         const unsigned* __restrict__ mx,
                         float* __restrict__ wbuf, float* __restrict__ den) {
    int e = blockIdx.x * blockDim.x + threadIdx.x;
    if (e >= ETOT) return;
    int d = (e < NE) ? dst[e] : (e - NE);
    float w = __expf(logit[e] - dec_ord(mx[d]));
    wbuf[e] = w;
    atomicAdd(den + d, w);
}
__global__ void k_edge_scatter(const int* __restrict__ src, const int* __restrict__ dst,
                               const float* __restrict__ wbuf, const float* __restrict__ den,
                               const float* __restrict__ xl, float* __restrict__ out) {
    int e = (blockIdx.x * blockDim.x + threadIdx.x) >> 5;
    int lane = threadIdx.x & 31;
    if (e >= ETOT) return;
    int s, d;
    if (e < NE) { s = src[e]; d = dst[e]; } else { s = d = e - NE; }
    float alpha = wbuf[e] / (den[d] + 1e-16f);
    int c = lane * 4;
    float4 lv = *(const float4*)(xl + (size_t)s * NDOUT + c);
    float* op = out + (size_t)d * NDOUT + c;
    atomicAdd(op + 0, alpha * lv.x);
    atomicAdd(op + 1, alpha * lv.y);
    atomicAdd(op + 2, alpha * lv.z);
    atomicAdd(op + 3, alpha * lv.w);
}

// GAT output + bias -> bf16 A matrix for the MHA in-projection
__global__ void k_cvt_bias_bf(const float* __restrict__ in, const float* __restrict__ b,
                              unsigned short* __restrict__ out) {
    int i = blockIdx.x * blockDim.x + threadIdx.x;  // over N*128
    out[i] = f2bf(in[i] + b[i & 127]);
}
// qh/kh f32 [N,128] -> bf16 [N, H, 32] (dh padded 16->32 with zeros), optional scale
__global__ void k_cvt_head(const float* __restrict__ in, unsigned short* __restrict__ out,
                           float scale) {
    int i = blockIdx.x * blockDim.x + threadIdx.x;  // over N*H*32
    int t = i & 31, h = (i >> 5) & 7, node = i >> 8;
    out[i] = (t < NDH) ? f2bf(in[node * NDOUT + h * NDH + t] * scale) : (unsigned short)0;
}
// vh f32 [N,128] -> vt bf16 [B, H, 16, S] (transposed so K runs contiguous for B-fragments)
__global__ void k_cvt_vt(const float* __restrict__ in, unsigned short* __restrict__ vt) {
    int i = blockIdx.x * blockDim.x + threadIdx.x;  // over N*128
    int node = i >> 7, c = i & 127;
    int b = node >> 9, s0 = node & 511;
    int h = c >> 4, n = c & 15;
    vt[(((size_t)b * NH + h) * NDH + n) * NS + s0] = f2bf(in[i]);
}

// ---------- per-(graph, head, 32-query-row tile) attention ----------
// dynamic LDS: sc f32[32][512] | pb bf16[32][512] | rs f32[32]
__global__ __launch_bounds__(256)
void k_attn(const unsigned short* __restrict__ qh, const unsigned short* __restrict__ kh,
            const unsigned short* __restrict__ vt, float* __restrict__ o) {
    extern __shared__ char smem[];
    float*          sc = (float*)smem;
    unsigned short* pb = (unsigned short*)(smem + 32 * 512 * 4);
    float*          rs = (float*)(smem + 32 * 512 * 4 + 32 * 512 * 2);

    const int bx = blockIdx.x;
    const int qt = bx & 15;
    const int h  = (bx >> 4) & 7;
    const int b  = bx >> 7;
    const int wave = threadIdx.x >> 5;
    const int lane = threadIdx.x & 31;
    const int half = lane >> 4;
    const int l16  = lane & 15;
    const int kb0  = half * 8;

    union { v16bf v; u16x8 hh[2]; } af, bfr;

    // phase 1: scores 32x512 (64 WMMA tiles over 8 waves; 1/sqrt(dh) folded into qh)
    for (int t = wave; t < 64; t += 8) {
        int rt = t & 1, ct = t >> 1;
        const unsigned short* ap =
            qh + ((size_t)(b * NS + qt * 32 + rt * 16 + l16) * NH + h) * 32;
        af.hh[0] = *(const u16x8*)(ap + half * 16);
        af.hh[1] = *(const u16x8*)(ap + half * 16 + 8);
        const unsigned short* bp =
            kh + ((size_t)(b * NS + ct * 16 + l16) * NH + h) * 32 + half * 16;
        bfr.hh[0] = *(const u16x8*)(bp);
        bfr.hh[1] = *(const u16x8*)(bp + 8);
        v8f c = {};
        c = __builtin_amdgcn_wmma_f32_16x16x32_bf16(false, af.v, false, bfr.v,
                                                    (short)0, c, false, false);
        int rb = rt * 16 + half * 8;
#pragma unroll
        for (int r = 0; r < 8; ++r) sc[(rb + r) * 512 + ct * 16 + l16] = c[r];
    }
    __syncthreads();

    // phase 2: row softmax -> unnormalized exp (bf16) + row sums
    if (threadIdx.x < 32) {
        int r = threadIdx.x;
        float m = -3.4e38f;
        for (int c = 0; c < 512; ++c) m = fmaxf(m, sc[r * 512 + c]);
        float s = 0.0f;
        for (int c = 0; c < 512; ++c) {
            float e = __expf(sc[r * 512 + c] - m);
            s += e;
            pb[r * 512 + c] = f2bf(e);
        }
        rs[r] = s;
    }
    __syncthreads();

    // phase 3: O = P @ V (two 16x16 tiles, K=512), A-fragments from LDS
    if (wave < 2) {
        v8f acc = {};
        const unsigned short* ap = pb + (size_t)(wave * 16 + l16) * 512;
        const unsigned short* bp =
            vt + (((size_t)b * NH + h) * NDH + l16) * NS + half * 16;
        for (int k0 = 0; k0 < 512; k0 += 32) {
            af.hh[0]  = *(const u16x8*)(ap + k0 + half * 16);
            af.hh[1]  = *(const u16x8*)(ap + k0 + half * 16 + 8);
            bfr.hh[0] = *(const u16x8*)(bp + k0);
            bfr.hh[1] = *(const u16x8*)(bp + k0 + 8);
            acc = __builtin_amdgcn_wmma_f32_16x16x32_bf16(false, af.v, false, bfr.v,
                                                          (short)0, acc, false, false);
        }
        int rb = wave * 16 + half * 8;
#pragma unroll
        for (int r = 0; r < 8; ++r) {
            int row = rb + r;
            int node = b * NS + qt * 32 + row;
            o[(size_t)node * NDOUT + h * NDH + l16] = acc[r] / rs[row];
        }
    }
}

// ============================================================================
extern "C" void kernel_launch(void* const* d_in, const int* in_sizes, int n_in,
                              void* d_out, int out_size, void* d_ws, size_t ws_size,
                              hipStream_t stream) {
    (void)in_sizes; (void)n_in; (void)out_size; (void)ws_size;

    const float* x    = (const float*)d_in[0];
    const int*   eidx = (const int*)  d_in[1];
    const int*   src  = eidx;
    const int*   dst  = eidx + NE;
    const float* ea   = (const float*)d_in[2];
    const int*   batch= (const int*)  d_in[3];

    const float* Wl_q = (const float*)d_in[4];  const float* bl_q = (const float*)d_in[5];
    const float* Wr_q = (const float*)d_in[6];  const float* br_q = (const float*)d_in[7];
    const float* We_q = (const float*)d_in[8];  const float* at_q = (const float*)d_in[9];
    const float* b_q  = (const float*)d_in[10];
    const float* Wl_k = (const float*)d_in[11]; const float* bl_k = (const float*)d_in[12];
    const float* Wr_k = (const float*)d_in[13]; const float* br_k = (const float*)d_in[14];
    const float* We_k = (const float*)d_in[15]; const float* at_k = (const float*)d_in[16];
    const float* b_k  = (const float*)d_in[17];
    const float* Wl_v = (const float*)d_in[18]; const float* bl_v = (const float*)d_in[19];
    const float* Wr_v = (const float*)d_in[20]; const float* br_v = (const float*)d_in[21];
    const float* We_v = (const float*)d_in[22]; const float* at_v = (const float*)d_in[23];
    const float* b_v  = (const float*)d_in[24];
    const float* W_in = (const float*)d_in[25]; const float* b_in = (const float*)d_in[26];
    const float* W_out= (const float*)d_in[27]; const float* b_out= (const float*)d_in[28];

    // ---- workspace carve (aligned 256B) ----
    char* w = (char*)d_ws;
    size_t off = 0;
    auto alloc = [&](size_t bytes) -> void* {
        off = (off + 255) & ~(size_t)255;
        void* p = w + off;
        off += bytes;
        return p;
    };
    float*          gc    = (float*)alloc((size_t)NB * NDIN * 4);
    unsigned short* xc    = (unsigned short*)alloc((size_t)NNODES * 256 * 2);
    unsigned short* xb    = (unsigned short*)alloc((size_t)NNODES * NDIN * 2);
    unsigned short* bt_lq = (unsigned short*)alloc(128 * 256 * 2);
    unsigned short* bt_rq = (unsigned short*)alloc(128 * 256 * 2);
    unsigned short* bt_lk = (unsigned short*)alloc(128 * 256 * 2);
    unsigned short* bt_rk = (unsigned short*)alloc(128 * 256 * 2);
    unsigned short* bt_lv = (unsigned short*)alloc(128 * 128 * 2);
    unsigned short* bt_rv = (unsigned short*)alloc(128 * 128 * 2);
    unsigned short* bt_in = (unsigned short*)alloc(384 * 128 * 2);
    unsigned short* bt_o  = (unsigned short*)alloc(128 * 128 * 2);
    float*          acc16 = (float*)alloc(NEDIM * 4);
    float*          epm   = (float*)alloc(NDOUT * 4);
    float*          xl    = (float*)alloc((size_t)NNODES * NDOUT * 4);
    float*          xr    = (float*)alloc((size_t)NNODES * NDOUT * 4);
    float*          logit = (float*)alloc((size_t)ETOT * 4);
    float*          wbuf  = (float*)alloc((size_t)ETOT * 4);
    unsigned*       mx    = (unsigned*)alloc((size_t)NNODES * 4);
    float*          den   = (float*)alloc((size_t)NNODES * 4);
    float*          gq    = (float*)alloc((size_t)NNODES * NDOUT * 4);
    float*          gk    = (float*)alloc((size_t)NNODES * NDOUT * 4);
    float*          gv    = (float*)alloc((size_t)NNODES * NDOUT * 4);
    unsigned short* aq    = (unsigned short*)alloc((size_t)NNODES * NDOUT * 2);
    unsigned short* ak    = (unsigned short*)alloc((size_t)NNODES * NDOUT * 2);
    unsigned short* av    = (unsigned short*)alloc((size_t)NNODES * NDOUT * 2);
    unsigned short* qhb   = (unsigned short*)alloc((size_t)NNODES * NH * 32 * 2);
    unsigned short* khb   = (unsigned short*)alloc((size_t)NNODES * NH * 32 * 2);
    unsigned short* vt    = (unsigned short*)alloc((size_t)NNODES * NDOUT * 2);
    // aliases (lifetimes disjoint): qh/kh/vh reuse GAT outputs; o/ob reuse xl/xr
    float*          qh = gq;  float* kh = gk;  float* vh = gv;
    float*          o  = xl;
    unsigned short* ob = (unsigned short*)xr;

    const dim3 B256(256);
    const int NELT = NNODES * NDOUT;

    // ---- stage 0: pooling, concatenation, weight conversion, edge-attr mean ----
    k_pool_max<<<NB, NDIN, 0, stream>>>(x, gc);
    k_build_xcat<<<(NNODES * 256) / 256, B256, 0, stream>>>(x, gc, batch, xc, xb);
    k_cvt_t<<<(128 * 256) / 256, B256, 0, stream>>>(Wl_q, bt_lq, 256, 128, 256);
    k_cvt_t<<<(128 * 256) / 256, B256, 0, stream>>>(Wr_q, bt_rq, 256, 128, 256);
    k_cvt_t<<<(128 * 256) / 256, B256, 0, stream>>>(Wl_k, bt_lk, 256, 128, 256);
    k_cvt_t<<<(128 * 256) / 256, B256, 0, stream>>>(Wr_k, bt_rk, 256, 128, 256);
    k_cvt_t<<<(128 * 128) / 256, B256, 0, stream>>>(Wl_v, bt_lv, 128, 128, 128);
    k_cvt_t<<<(128 * 128) / 256, B256, 0, stream>>>(Wr_v, bt_rv, 128, 128, 128);
    k_cvt_bf<<<(384 * 128) / 256, B256, 0, stream>>>(W_in, bt_in, 384 * 128);   // rows = out-neurons: already Bt layout
    k_cvt_bf<<<(128 * 128) / 256, B256, 0, stream>>>(W_out, bt_o, 128 * 128);
    k_zero_f<<<1, NEDIM, 0, stream>>>(acc16, NEDIM);
    k_ea_sum<<<1024, B256, 0, stream>>>(ea, acc16);

    // ---- per-GATv2 pipeline ----
    auto run_gat = [&](const unsigned short* Abf, int K,
                       const unsigned short* btl, const unsigned short* btr,
                       const float* bl, const float* br,
                       const float* We, const float* att, float* gout) {
        dim3 gg(NNODES / 16);
        k_gemm_bf16<<<gg, B256, 0, stream>>>(Abf, K, btl, K, bl, xl, NDOUT, K);
        k_gemm_bf16<<<gg, B256, 0, stream>>>(Abf, K, btr, K, br, xr, NDOUT, K);
        k_eproj_mean<<<1, NDOUT, 0, stream>>>(acc16, We, epm);
        k_zero_u<<<NNODES / 256, B256, 0, stream>>>(mx, NNODES);
        k_zero_f<<<NNODES / 256, B256, 0, stream>>>(den, NNODES);
        k_zero_f<<<NELT / 256, B256, 0, stream>>>(gout, NELT);
        k_edge_logits<<<ETOT / 8, B256, 0, stream>>>(src, dst, ea, We, epm, xl, xr, att, logit, mx);
        k_edge_w<<<(ETOT + 255) / 256, B256, 0, stream>>>(dst, logit, mx, wbuf, den);
        k_edge_scatter<<<ETOT / 8, B256, 0, stream>>>(src, dst, wbuf, den, xl, gout);
    };
    run_gat(xc, 256, bt_lq, bt_rq, bl_q, br_q, We_q, at_q, gq);
    k_cvt_bias_bf<<<NELT / 256, B256, 0, stream>>>(gq, b_q, aq);
    run_gat(xc, 256, bt_lk, bt_rk, bl_k, br_k, We_k, at_k, gk);
    k_cvt_bias_bf<<<NELT / 256, B256, 0, stream>>>(gk, b_k, ak);
    run_gat(xb, 128, bt_lv, bt_rv, bl_v, br_v, We_v, at_v, gv);
    k_cvt_bias_bf<<<NELT / 256, B256, 0, stream>>>(gv, b_v, av);

    // ---- MHA in-projection (q/k/v), head repack, attention, out-projection ----
    dim3 gg(NNODES / 16);
    k_gemm_bf16<<<gg, B256, 0, stream>>>(aq, 128, bt_in,             128, b_in,       qh, NDOUT, 128);
    k_gemm_bf16<<<gg, B256, 0, stream>>>(ak, 128, bt_in + 128 * 128, 128, b_in + 128, kh, NDOUT, 128);
    k_gemm_bf16<<<gg, B256, 0, stream>>>(av, 128, bt_in + 256 * 128, 128, b_in + 256, vh, NDOUT, 128);
    k_cvt_head<<<(NNODES * NH * 32) / 256, B256, 0, stream>>>(qh, qhb, 0.25f);  // 1/sqrt(16)
    k_cvt_head<<<(NNODES * NH * 32) / 256, B256, 0, stream>>>(kh, khb, 1.0f);
    k_cvt_vt<<<NELT / 256, B256, 0, stream>>>(vh, vt);

    size_t shbytes = 32 * 512 * 4 + 32 * 512 * 2 + 32 * 4;   // 96 KB + 128 B (< 320 KB/WGP)
    k_attn<<<NB * NH * (NS / 32), B256, shbytes, stream>>>(qhb, khb, vt, o);

    k_cvt_bf<<<NELT / 256, B256, 0, stream>>>(o, ob, NELT);
    k_gemm_bf16<<<gg, B256, 0, stream>>>(ob, 128, bt_o, 128, b_out, (float*)d_out, NDOUT, 128);
}
